// TaskAlignedAssigner_20564303414112
// MI455X (gfx1250) — compile-verified, matrix operand order checked by reading, not run on hardware
//
#include <hip/hip_runtime.h>
#include <hip/hip_bf16.h>

// Problem constants fixed by the reference's setup_inputs().
#define BS   8
#define NA   8400     // anchors
#define NM   128      // max gt boxes
#define NC   80       // classes
#define TOPKN 13
#define EPSF 1e-9f

typedef __attribute__((ext_vector_type(8))) int v8i;

// ---------------------------------------------------------------------------
// Kernel Z: zero the int8 candidate matrix (bs*M*A bytes) with 16B stores.
// ---------------------------------------------------------------------------
__global__ __launch_bounds__(256) void zero_kernel(int4* __restrict__ p, int n4) {
    int i = blockIdx.x * 256 + threadIdx.x;
    if (i < n4) p[i] = make_int4(0, 0, 0, 0);
}

// ---------------------------------------------------------------------------
// Kernel A: per-anchor argmax over NC classes -> float(class index).
// First-occurrence max to match jnp.argmax tie rule.
// ---------------------------------------------------------------------------
__global__ __launch_bounds__(256) void argmax_cls_kernel(const float* __restrict__ sc,
                                                         float* __restrict__ cls_f) {
    int i = blockIdx.x * 256 + threadIdx.x;
    if (i >= BS * NA) return;
    const float* p = sc + (size_t)i * NC;
    float bv = p[0];
    int bi = 0;
#pragma unroll 4
    for (int c = 1; c < NC; ++c) {
        float v = p[c];
        if (v > bv) { bv = v; bi = c; }
    }
    cls_f[i] = (float)bi;
}

// ---------------------------------------------------------------------------
// Kernel B: one workgroup per (b, m) gt row.  Computes align on the fly and
// selects top-13 anchors (value desc, index asc ties — matches lax.top_k),
// then scatters 1-bytes into cand[b][m][:].
// ---------------------------------------------------------------------------
__global__ __launch_bounds__(256) void topk_kernel(const float* __restrict__ pd_boxes,
                                                   const float* __restrict__ cls_f,
                                                   const float* __restrict__ gt_boxes,
                                                   const float* __restrict__ mask_gt,
                                                   unsigned char* __restrict__ cand) {
    const int bm  = blockIdx.x;          // 0 .. BS*NM-1
    const int b   = bm / NM;
    const int tid = threadIdx.x;

    const float4 g   = reinterpret_cast<const float4*>(gt_boxes)[bm];
    const float  mgt = mask_gt[bm];
    const float  ag  = (g.z - g.x) * (g.w - g.y);

    float vals[TOPKN];
    int   idxs[TOPKN];
#pragma unroll
    for (int k = 0; k < TOPKN; ++k) { vals[k] = -1.0f; idxs[k] = 0x7FFFFFFF; }

    const float4* pb4 = reinterpret_cast<const float4*>(pd_boxes) + (size_t)b * NA;
    const float*  cf  = cls_f + (size_t)b * NA;

    for (int a = tid; a < NA; a += 256) {
        if (a + 256 < NA) __builtin_prefetch(&pb4[a + 256], 0, 1); // global_prefetch_b8
        const float4 p = pb4[a];
        // pairwise IoU (xyxy)
        float ltx = fmaxf(g.x, p.x), lty = fmaxf(g.y, p.y);
        float rbx = fminf(g.z, p.z), rby = fminf(g.w, p.w);
        float w = fmaxf(rbx - ltx, 0.0f), h = fmaxf(rby - lty, 0.0f);
        float inter = w * h;
        float ap    = (p.z - p.x) * (p.w - p.y);
        float iou   = inter / (ag + ap - inter + EPSF);
        // center-inside-gt test
        float cx = (p.x + p.z) * 0.5f, cy = (p.y + p.w) * 0.5f;
        float dmin = fminf(fminf(cx - g.x, g.z - cx), fminf(cy - g.y, g.w - cy));
        float ing  = (dmin > EPSF) ? 1.0f : 0.0f;
        // align = cls^1 * iou^6 * in_gts * mask_gt
        float i3 = iou * iou * iou;
        float v  = cf[a] * (i3 * i3) * ing * mgt;
        int   vi = a;
        // fully-predicated register bubble insert (keeps arrays in VGPRs)
#pragma unroll
        for (int k = 0; k < TOPKN; ++k) {
            bool bt = (v > vals[k]) || (v == vals[k] && vi < idxs[k]);
            float tv = vals[k]; int ti = idxs[k];
            if (bt) { vals[k] = v; idxs[k] = vi; v = tv; vi = ti; }
        }
    }

    __shared__ float sv[256 * TOPKN];
    __shared__ int   si[256 * TOPKN];
#pragma unroll
    for (int k = 0; k < TOPKN; ++k) {
        sv[tid * TOPKN + k] = vals[k];
        si[tid * TOPKN + k] = idxs[k];
    }
    __syncthreads();

    if (tid == 0) {
        float mv[TOPKN]; int mi[TOPKN];
#pragma unroll
        for (int k = 0; k < TOPKN; ++k) { mv[k] = -1.0f; mi[k] = 0x7FFFFFFF; }
        for (int e = 0; e < 256 * TOPKN; ++e) {
            float v = sv[e]; int vi = si[e];
            if (!((v > mv[TOPKN - 1]) || (v == mv[TOPKN - 1] && vi < mi[TOPKN - 1]))) continue;
#pragma unroll
            for (int k = 0; k < TOPKN; ++k) {
                bool bt = (v > mv[k]) || (v == mv[k] && vi < mi[k]);
                float tv = mv[k]; int ti = mi[k];
                if (bt) { mv[k] = v; mi[k] = vi; v = tv; vi = ti; }
            }
        }
        unsigned char* crow = cand + (size_t)bm * NA;
#pragma unroll
        for (int k = 0; k < TOPKN; ++k)
            if (mi[k] < NA) crow[mi[k]] = 1;
    }
}

// ---------------------------------------------------------------------------
// Kernel C: counts[b][a] = sum over M of cand[b][m][a], computed as the int8
// matmul ones(16x64) x cand(64x16) on V_WMMA_I32_16X16X64_IU8 (2 K-steps
// cover M=128).  One wave32 handles one 16-anchor tile; with all-ones A every
// output row holds the column sums, so lanes 0-15 read D row 0.
// ---------------------------------------------------------------------------
__global__ __launch_bounds__(256) void count_wmma_kernel(const unsigned char* __restrict__ cand,
                                                         int* __restrict__ counts) {
    const int lane = threadIdx.x & 31;
    const int wave = threadIdx.x >> 5;
    const int tile = blockIdx.x * 8 + wave;          // BS*(NA/16) tiles total
    const int b    = tile / (NA / 16);
    const int a0   = (tile % (NA / 16)) * 16;
    const int col  = lane & 15;
    const int grp  = lane >> 4;                      // half-wave K group

    v8i ones, acc;
#pragma unroll
    for (int v = 0; v < 8; ++v) { ones[v] = 0x01010101; acc[v] = 0; }

    const unsigned char* base = cand + (size_t)b * NM * NA + a0 + col;
#pragma unroll
    for (int w = 0; w < 2; ++w) {                    // K = 0..63, 64..127
        v8i bmat;
#pragma unroll
        for (int v = 0; v < 8; ++v) {
            unsigned int packed = 0;
#pragma unroll
            for (int j = 0; j < 4; ++j) {
                // 8-bit B 64x16 layout: V0..3 -> K[grp*16 .. grp*16+15],
                //                       V4..7 -> K[32+grp*16 .. 32+grp*16+15]
                int kl = (v < 4) ? (grp * 16 + v * 4 + j)
                                 : (32 + grp * 16 + (v - 4) * 4 + j);
                int m  = w * 64 + kl;
                packed |= ((unsigned int)base[(size_t)m * NA]) << (8 * j);
            }
            bmat[v] = (int)packed;
        }
        acc = __builtin_amdgcn_wmma_i32_16x16x64_iu8(false, ones, false, bmat,
                                                     acc, false, false);
    }
    if (lane < 16) counts[(size_t)b * NA + a0 + col] = acc[0];
}

// ---------------------------------------------------------------------------
// Kernel D: per-anchor resolution + targets (tl, tb, mask).
// count==1 -> the single assigned gt; count>1 -> argmax raw IoU over all gts
// (matches reference best_g over unmasked overlaps); count==0 -> bg (tl=80,
// tb still = gt_boxes[b][0], exactly like the reference gather with am=0).
// ---------------------------------------------------------------------------
__global__ __launch_bounds__(256) void resolve_kernel(const float* __restrict__ pd_boxes,
                                                      const float* __restrict__ gt_boxes,
                                                      const int* __restrict__ gt_labels,
                                                      const unsigned char* __restrict__ cand,
                                                      const int* __restrict__ counts,
                                                      float* __restrict__ out) {
    __shared__ float4 sgt[NM];
    __shared__ int    sgl[NM];
    const int bpb = (NA + 255) / 256;                // blocks per batch
    const int b   = blockIdx.x / bpb;
    const int a   = (blockIdx.x % bpb) * 256 + threadIdx.x;

    if (threadIdx.x < NM) {
        const float* gp = gt_boxes + ((size_t)b * NM + threadIdx.x) * 4;
        sgt[threadIdx.x] = make_float4(gp[0], gp[1], gp[2], gp[3]);
        sgl[threadIdx.x] = gt_labels[(size_t)b * NM + threadIdx.x];
    }
    __syncthreads();
    if (a >= NA) return;

    const int cnt = counts[(size_t)b * NA + a];
    int am = 0;
    if (cnt == 1) {
        const unsigned char* cc = cand + (size_t)b * NM * NA + a;
        for (int m = 0; m < NM; ++m) {
            if (cc[(size_t)m * NA] != 0) { am = m; break; }
        }
    } else if (cnt > 1) {
        const float4 p  = reinterpret_cast<const float4*>(pd_boxes)[(size_t)b * NA + a];
        const float  ap = (p.z - p.x) * (p.w - p.y);
        float bestv = -1.0f;
        for (int m = 0; m < NM; ++m) {
            float4 g = sgt[m];
            float ltx = fmaxf(g.x, p.x), lty = fmaxf(g.y, p.y);
            float rbx = fminf(g.z, p.z), rby = fminf(g.w, p.w);
            float w = fmaxf(rbx - ltx, 0.0f), h = fmaxf(rby - lty, 0.0f);
            float inter = w * h;
            float ag    = (g.z - g.x) * (g.w - g.y);
            float iou   = inter / (ag + ap - inter + EPSF);
            if (iou > bestv) { bestv = iou; am = m; }   // first-max tie rule
        }
    }

    const int  idx = b * NA + a;
    const bool fg  = cnt > 0;
    const int  tl  = fg ? sgl[am] : NC;
    out[idx] = (float)tl;                                     // tl  @ 0
    reinterpret_cast<float4*>(out + BS * NA)[idx] = sgt[am];  // tb  @ BS*NA
    out[(size_t)BS * NA * 85 + idx] = fg ? 1.0f : 0.0f;       // mask@ BS*NA*85
}

// ---------------------------------------------------------------------------
// Kernel E: ts one-hot (bs, A, 80) from tl already written into d_out.
// tl==80 (background) -> all zeros, matching one_hot(81)[...,:80].
// ---------------------------------------------------------------------------
__global__ __launch_bounds__(256) void onehot_kernel(float* __restrict__ out) {
    const int i = blockIdx.x * 256 + threadIdx.x;   // < BS*NA*NC
    if (i >= BS * NA * NC) return;
    const int c  = i % NC;
    const int ba = i / NC;
    const int tl = (int)out[ba];
    out[(size_t)BS * NA * 5 + i] = (c == tl) ? 1.0f : 0.0f;   // ts @ BS*NA*5
}

// ---------------------------------------------------------------------------
extern "C" void kernel_launch(void* const* d_in, const int* in_sizes, int n_in,
                              void* d_out, int out_size, void* d_ws, size_t ws_size,
                              hipStream_t stream) {
    (void)in_sizes; (void)n_in; (void)out_size; (void)ws_size;
    const float* pd_scores = (const float*)d_in[0];
    const float* pd_boxes  = (const float*)d_in[1];
    const int*   gt_labels = (const int*)d_in[2];
    const float* gt_boxes  = (const float*)d_in[3];
    const float* mask_gt   = (const float*)d_in[4];
    float* out = (float*)d_out;

    // workspace layout (all 16B aligned)
    unsigned char* cand  = (unsigned char*)d_ws;                                // BS*NM*NA bytes
    float*         cls_f = (float*)((char*)d_ws + (size_t)BS * NM * NA);        // BS*NA floats
    int*           cnts  = (int*)((char*)d_ws + (size_t)BS * NM * NA
                                             + (size_t)BS * NA * sizeof(float)); // BS*NA ints

    const int n4 = (BS * NM * NA) / 16;
    zero_kernel<<<(n4 + 255) / 256, 256, 0, stream>>>((int4*)cand, n4);

    argmax_cls_kernel<<<(BS * NA + 255) / 256, 256, 0, stream>>>(pd_scores, cls_f);

    topk_kernel<<<BS * NM, 256, 0, stream>>>(pd_boxes, cls_f, gt_boxes, mask_gt, cand);

    count_wmma_kernel<<<(BS * (NA / 16)) / 8, 256, 0, stream>>>(cand, cnts);

    resolve_kernel<<<BS * ((NA + 255) / 256), 256, 0, stream>>>(pd_boxes, gt_boxes,
                                                                gt_labels, cand, cnts, out);

    onehot_kernel<<<(BS * NA * NC + 255) / 256, 256, 0, stream>>>(out);
}
